// MoELayer_27960237097214
// MI455X (gfx1250) — matrix operand order repaired and checked by reading, not compiled, not run on hardware
//
#include <hip/hip_runtime.h>
#include <stdint.h>

#define D_MODEL 1024
#define D_FF    4096
#define N_EXP   8
#define T_TOK   8192   // 4 * 2048

#define XROW_LDS 1032  // 1024 + 8 pad elems -> 516-dword row stride (bank-conflict free)
#define HROW_LDS 72    // 64 + 8 pad elems   -> 36-dword row stride  (bank-conflict free)

typedef __attribute__((ext_vector_type(16))) __bf16 v16bf;
typedef __attribute__((ext_vector_type(8)))  __bf16 v8bf;
typedef __attribute__((ext_vector_type(8)))  float  v8f;

#define WMMA_BF16(a, b, c) \
  __builtin_amdgcn_wmma_f32_16x16x32_bf16(false, (a), false, (b), (short)0, (c), false, false)

// 16-bit A fragment (16x32, M = lane%16): lanes 0-15 hold K k0..k0+15,
// lanes 16-31 hold K k0+16..k0+31  (matches B fragment order)
__device__ __forceinline__ v16bf load_a_frag(const __bf16* rowp, int k0, int lhalf) {
  union { v16bf v; v8bf h[2]; } u;
  u.h[0] = *(const v8bf*)(rowp + k0 + lhalf * 16);
  u.h[1] = *(const v8bf*)(rowp + k0 + lhalf * 16 + 8);
  return u.v;
}

// 16-bit B fragment (32x16, N = lane%16): lanes 0-15 hold K k0..k0+15, lanes 16-31 K k0+16..k0+31.
// colp points at row N of the transposed weight (contiguous over K).
__device__ __forceinline__ v16bf load_b_frag(const __bf16* colp, int k0, int lhalf) {
  return *(const v16bf*)(colp + k0 + lhalf * 16);
}

// ---------------- init: zero out[] and expert counters ----------------
__global__ void init_kernel(float* __restrict__ out, int* __restrict__ counts) {
  size_t i = (size_t)blockIdx.x * blockDim.x + threadIdx.x;
  size_t n4 = (size_t)T_TOK * D_MODEL / 4;
  if (i < n4) ((float4*)out)[i] = make_float4(0.f, 0.f, 0.f, 0.f);
  if (i < N_EXP) counts[i] = 0;
}

// ---------------- gating: one wave32 per token ----------------
__global__ void gate_kernel(const float* __restrict__ x, const float* __restrict__ Wg,
                            const float* __restrict__ bg, float* __restrict__ gate_out,
                            float* __restrict__ combine) {
  int gid  = blockIdx.x * blockDim.x + threadIdx.x;
  int t    = gid >> 5;
  int lane = gid & 31;
  if (t >= T_TOK) return;
  const float* xr = x + (size_t)t * D_MODEL;
  float acc[8] = {0.f, 0.f, 0.f, 0.f, 0.f, 0.f, 0.f, 0.f};
  for (int d = lane; d < D_MODEL; d += 32) {
    float xv = xr[d];
    const float4* w4 = (const float4*)(Wg + d * 8);
    float4 w0 = w4[0], w1 = w4[1];
    acc[0] += xv * w0.x; acc[1] += xv * w0.y; acc[2] += xv * w0.z; acc[3] += xv * w0.w;
    acc[4] += xv * w1.x; acc[5] += xv * w1.y; acc[6] += xv * w1.z; acc[7] += xv * w1.w;
  }
#pragma unroll
  for (int off = 16; off > 0; off >>= 1)
#pragma unroll
    for (int e = 0; e < 8; e++) acc[e] += __shfl_xor(acc[e], off, 32);
  if (lane == 0) {
    float lg[8], p[8];
    float m = -1e30f;
#pragma unroll
    for (int e = 0; e < 8; e++) { lg[e] = acc[e] + bg[e]; m = fmaxf(m, lg[e]); }
    float s = 0.f;
#pragma unroll
    for (int e = 0; e < 8; e++) { p[e] = expf(lg[e] - m); s += p[e]; }
    float inv = 1.f / s;
#pragma unroll
    for (int e = 0; e < 8; e++) p[e] *= inv;
    int i1 = 0;
#pragma unroll
    for (int e = 1; e < 8; e++) if (p[e] > p[i1]) i1 = e;          // first index wins ties
    int i2 = (i1 == 0) ? 1 : 0;
#pragma unroll
    for (int e = 0; e < 8; e++) if (e != i1 && p[e] > p[i2]) i2 = e;
#pragma unroll
    for (int e = 0; e < 8; e++) {
      gate_out[(size_t)t * 8 + e] = p[e];
      combine[(size_t)t * 8 + e]  = (e == i1) ? p[i1] : ((e == i2) ? p[i2] : 0.f);
    }
  }
}

// ---------------- routing: build per-expert token lists ----------------
__global__ void route_kernel(const float* __restrict__ combine, int* __restrict__ counts,
                             int* __restrict__ lists) {
  int t = blockIdx.x * blockDim.x + threadIdx.x;
  if (t >= T_TOK) return;
#pragma unroll
  for (int e = 0; e < 8; e++) {
    if (combine[(size_t)t * 8 + e] > 0.f) {
      int p = atomicAdd(&counts[e], 1);
      lists[(size_t)e * T_TOK + p] = t;
    }
  }
}

// ---------------- fp32 -> bf16 converts (weights transposed) ----------------
__global__ void cvt_x_kernel(const float* __restrict__ x, __bf16* __restrict__ xb) {
  size_t i = (size_t)blockIdx.x * blockDim.x + threadIdx.x;
  if (i < (size_t)T_TOK * D_MODEL) xb[i] = (__bf16)x[i];
}
// W1[e][d][f] -> W1T[e][f][d]
__global__ void cvt_w1t_kernel(const float* __restrict__ w1, __bf16* __restrict__ w1t) {
  size_t i = (size_t)blockIdx.x * blockDim.x + threadIdx.x;
  if (i >= (size_t)N_EXP * D_FF * D_MODEL) return;
  int d = (int)(i % D_MODEL);
  size_t r = i / D_MODEL;
  int f = (int)(r % D_FF);
  int e = (int)(r / D_FF);
  w1t[i] = (__bf16)w1[((size_t)e * D_MODEL + d) * D_FF + f];
}
// W2[e][f][n] -> W2T[e][n][f]
__global__ void cvt_w2t_kernel(const float* __restrict__ w2, __bf16* __restrict__ w2t) {
  size_t i = (size_t)blockIdx.x * blockDim.x + threadIdx.x;
  if (i >= (size_t)N_EXP * D_MODEL * D_FF) return;
  int f = (int)(i % D_FF);
  size_t r = i / D_FF;
  int n = (int)(r % D_MODEL);
  int e = (int)(r / D_MODEL);
  w2t[i] = (__bf16)w2[((size_t)e * D_FF + f) * D_MODEL + n];
}

// ---------------- fused expert FFN: 32-token tile, full 1024 output cols ----------------
__global__ __launch_bounds__(256, 1)
void ffn_kernel(const __bf16* __restrict__ xb, const __bf16* __restrict__ w1t,
                const __bf16* __restrict__ w2t, const float* __restrict__ b1,
                const float* __restrict__ b2, const float* __restrict__ combine,
                const int* __restrict__ counts, const int* __restrict__ lists,
                float* __restrict__ out) {
  __shared__ int tile_tok[32];
  __shared__ __bf16 lds_x[32 * XROW_LDS];   // gathered x tile, padded rows (~66 KB)
  __shared__ __bf16 lds_h[32 * HROW_LDS];   // relu'd hidden chunk, padded rows (~4.5 KB)

  int cnt_l[8];
#pragma unroll
  for (int i = 0; i < 8; i++) cnt_l[i] = counts[i];

  int tile = blockIdx.x;
  int e = -1, tbase = 0, run = 0;
#pragma unroll
  for (int i = 0; i < 8; i++) {
    int nt = (cnt_l[i] + 31) >> 5;
    if (e < 0 && tile < run + nt) { e = i; tbase = (tile - run) * 32; }
    run += nt;
  }
  if (e < 0) return;
  int cnt = cnt_l[e];

  if (threadIdx.x < 32) {
    int idx = tbase + (int)threadIdx.x;
    if (idx >= cnt) idx = cnt - 1;                  // pad rows clamp (writeback guarded)
    tile_tok[threadIdx.x] = lists[(size_t)e * T_TOK + idx];
  }
  __syncthreads();

  // ---- stage gathered x tile into LDS once (32 rows x 1024 bf16, 16B chunks) ----
  for (int c = threadIdx.x; c < 32 * (D_MODEL / 8); c += 256) {
    int row = c >> 7;                  // /128 chunks per row
    int off = (c & 127) * 8;
    int tok = tile_tok[row];
    *(v8bf*)(lds_x + (size_t)row * XROW_LDS + off) =
        *(const v8bf*)(xb + (size_t)tok * D_MODEL + off);
  }
  __syncthreads();

  const int w     = threadIdx.x >> 5;
  const int lane  = threadIdx.x & 31;
  const int lhalf = lane >> 4;
  const int l15   = lane & 15;
  const int mt1   = w >> 2;    // GEMM1: this wave's 16-row M tile (0..1)
  const int nt1   = w & 3;     // GEMM1: this wave's 16-col ff tile within chunk (0..3)

  const __bf16* xrowA = lds_x + (size_t)(mt1 * 16 + l15) * XROW_LDS;

  v8f zero = {0.f, 0.f, 0.f, 0.f, 0.f, 0.f, 0.f, 0.f};
  v8f acc2[2][8];
#pragma unroll
  for (int mt = 0; mt < 2; mt++)
#pragma unroll
    for (int nt = 0; nt < 8; nt++) acc2[mt][nt] = zero;

#pragma unroll 1
  for (int c0 = 0; c0 < D_FF; c0 += 64) {
    // ---- GEMM1: h(32x64) = relu(x_tile @ W1[:, c0:c0+64] + b1), one 16x16 tile/wave ----
    int ffcol = c0 + nt1 * 16 + l15;                       // B column N = lane%16
    const __bf16* w1col = w1t + ((size_t)e * D_FF + ffcol) * D_MODEL;
    v8f hacc = zero;
#pragma unroll 4
    for (int k0 = 0; k0 < D_MODEL; k0 += 32) {
      v16bf a = load_a_frag(xrowA, k0, lhalf);             // ds_load_b128 x2 from LDS
      v16bf b = load_b_frag(w1col, k0, lhalf);             // global 32B from L2-resident W1T
      hacc = WMMA_BF16(a, b, hacc);
    }
    float bias1 = b1[(size_t)e * D_FF + ffcol];            // same N for all 8 D-VGPRs

    __syncthreads();                                       // prior GEMM2 done with lds_h
    {
      int hm = mt1 * 16 + lhalf * 8;                       // D layout: M = r + 8*(lane>=16)
      int hn = nt1 * 16 + l15;
#pragma unroll
      for (int r = 0; r < 8; r++) {
        float v = hacc[r] + bias1;
        v = v > 0.f ? v : 0.f;
        lds_h[(hm + r) * HROW_LDS + hn] = (__bf16)v;       // hw f32->bf16 convert
      }
    }
    __syncthreads();

    // ---- GEMM2: y(32x1024) += h(32x64) @ W2[c0:c0+64, :], 128 cols/wave ----
#pragma unroll
    for (int kk = 0; kk < 64; kk += 32) {
      v16bf a0 = load_a_frag(lds_h + (size_t)(0 + l15) * HROW_LDS, kk, lhalf);
      v16bf a1 = load_a_frag(lds_h + (size_t)(16 + l15) * HROW_LDS, kk, lhalf);
#pragma unroll
      for (int nt = 0; nt < 8; nt++) {
        int n = w * 128 + nt * 16 + l15;
        const __bf16* w2col = w2t + ((size_t)e * D_MODEL + n) * D_FF + c0;
        v16bf b = load_b_frag(w2col, kk, lhalf);
        acc2[0][nt] = WMMA_BF16(a0, b, acc2[0][nt]);
        acc2[1][nt] = WMMA_BF16(a1, b, acc2[1][nt]);
      }
    }
  }

  // ---- scaled scatter-add: out[tok] += combine[tok,e] * (y + b2[e]) ----
#pragma unroll
  for (int mt = 0; mt < 2; mt++) {
#pragma unroll
    for (int r = 0; r < 8; r++) {
      int row = mt * 16 + lhalf * 8 + r;
      if (tbase + row < cnt) {
        int tok = tile_tok[row];
        float cw = combine[(size_t)tok * 8 + e];
#pragma unroll
        for (int nt = 0; nt < 8; nt++) {
          int col = w * 128 + nt * 16 + l15;
          float val = cw * (acc2[mt][nt][r] + b2[(size_t)e * D_MODEL + col]);
          atomicAdd(out + (size_t)tok * D_MODEL + col, val);
        }
      }
    }
  }
}

extern "C" void kernel_launch(void* const* d_in, const int* in_sizes, int n_in,
                              void* d_out, int out_size, void* d_ws, size_t ws_size,
                              hipStream_t stream) {
  (void)in_sizes; (void)n_in; (void)out_size; (void)ws_size;
  const float* x  = (const float*)d_in[0];
  const float* W1 = (const float*)d_in[1];
  const float* b1 = (const float*)d_in[2];
  const float* W2 = (const float*)d_in[3];
  const float* b2 = (const float*)d_in[4];
  const float* Wg = (const float*)d_in[5];
  const float* bg = (const float*)d_in[6];

  float* out      = (float*)d_out;
  float* gate_out = out + (size_t)T_TOK * D_MODEL;   // tuple part 2: (B,S,E)

  // workspace carve (all 256B-aligned chunks); total ~151.5 MB
  char* ws = (char*)d_ws;
  float* combine = (float*)ws;  ws += (size_t)T_TOK * 8 * sizeof(float);
  int* counts    = (int*)ws;    ws += 256;
  int* lists     = (int*)ws;    ws += (size_t)N_EXP * T_TOK * sizeof(int);
  __bf16* xb     = (__bf16*)ws; ws += (size_t)T_TOK * D_MODEL * 2;
  __bf16* w1t    = (__bf16*)ws; ws += (size_t)N_EXP * D_FF * D_MODEL * 2;
  __bf16* w2t    = (__bf16*)ws; ws += (size_t)N_EXP * D_MODEL * D_FF * 2;

  {
    int blocks = (int)(((size_t)T_TOK * D_MODEL / 4 + 255) / 256);
    init_kernel<<<blocks, 256, 0, stream>>>(out, counts);
  }
  gate_kernel<<<(T_TOK * 32 + 255) / 256, 256, 0, stream>>>(x, Wg, bg, gate_out, combine);
  route_kernel<<<(T_TOK + 255) / 256, 256, 0, stream>>>(combine, counts, lists);
  {
    int bx = (int)(((size_t)T_TOK * D_MODEL + 255) / 256);
    cvt_x_kernel<<<bx, 256, 0, stream>>>(x, xb);
    int bw = (int)(((size_t)N_EXP * D_FF * D_MODEL + 255) / 256);
    cvt_w1t_kernel<<<bw, 256, 0, stream>>>(W1, w1t);
    cvt_w2t_kernel<<<bw, 256, 0, stream>>>(W2, w2t);
  }
  // max tiles = sum_e ceil(cnt_e/32) <= 2*T/32 + 8 = 520 (every token routes to exactly 2 experts)
  ffn_kernel<<<8 + (2 * T_TOK) / 32, 256, 0, stream>>>(xb, w1t, w2t, b1, b2, combine,
                                                       counts, lists, out);
}
